// DirectSoftTreeEnsemble_43413529428399
// MI455X (gfx1250) — compile-verified, matrix operand order checked by reading, not run on hardware
//
#include <hip/hip_runtime.h>
#include <hip/hip_bf16.h>
#include <math.h>

typedef __attribute__((ext_vector_type(16))) _Float16 v16h;
typedef __attribute__((ext_vector_type(8)))  _Float16 v8h;
typedef __attribute__((ext_vector_type(8)))  float    v8f;

#define Bn 4096
#define Dn 1024
#define Cn 1000
#define Tn 32
#define NLEAF 64
#define NINT 63
#define KMU 2048      // T*64 flattened (t,leaf) contraction dim
#define CPAD 1024     // padded C for 64-wide N tiles

// ---------------------------------------------------------------------------
// Fragment loader: 16-bit A/B fragment for v_wmma_*_16x16x32_f16 (wave32).
// Lane L holds row/col (L&15); K-chunks: [k0+8*(L>>4),+8) and [k0+16+8*(L>>4),+8).
// Each chunk is one aligned 16B load.
// ---------------------------------------------------------------------------
__device__ __forceinline__ v16h load_frag16(const _Float16* rowp, int k0, int hs) {
    union { v16h v; v8h h[2]; } u;
    u.h[0] = *reinterpret_cast<const v8h*>(rowp + k0 + hs * 8);
    u.h[1] = *reinterpret_cast<const v8h*>(rowp + k0 + 16 + hs * 8);
    return u.v;
}

__device__ __forceinline__ v8f wmma_f16(v16h a, v16h b, v8f c) {
    return __builtin_amdgcn_wmma_f32_16x16x32_f16(false, a, false, b,
                                                  (short)0, c, false, false);
}

// ---------------------------------------------------------------------------
// Prep: f32 -> f16 conversions
// ---------------------------------------------------------------------------
__global__ void cvt_x_kernel(const float* __restrict__ x, _Float16* __restrict__ xh) {
    int i = blockIdx.x * 256 + threadIdx.x;           // grid covers Bn*Dn exactly
    xh[i] = (_Float16)x[i];
}

__global__ void cvt_w_kernel(const float* __restrict__ sw, _Float16* __restrict__ swh) {
    int e = blockIdx.x * 256 + threadIdx.x;           // grid covers Tn*64*Dn exactly
    int t = e >> 16;                                  // 64*1024 = 65536 per tree
    int rem = e & 65535;
    int i = rem >> 10;                                // node 0..63
    int d = rem & 1023;
    float v = (i < NINT) ? sw[((size_t)t * NINT + i) * Dn + d] : 0.0f;
    swh[e] = (_Float16)v;
}

// ---------------------------------------------------------------------------
// Leaf softmax -> transposed f16 matrix Pt[c, k], k = t*64 + l, rows c 0..1023
// (rows >= 1000 zeroed). One block per k.
// ---------------------------------------------------------------------------
__global__ void leaf_kernel(const float* __restrict__ leaf_logits,
                            const float* __restrict__ logT,
                            _Float16* __restrict__ Pt) {
    __shared__ float red[256];
    int k = blockIdx.x;
    int tid = threadIdx.x;
    const float* row = leaf_logits + (size_t)k * Cn;
    float temp = fminf(fmaxf(expf(logT[0]), 0.1f), 5.0f);
    float invT = 1.0f / temp;

    float mx = -INFINITY;
    for (int c = tid; c < Cn; c += 256) mx = fmaxf(mx, row[c] * invT);
    red[tid] = mx;
    __syncthreads();
    for (int s = 128; s > 0; s >>= 1) {
        if (tid < s) red[tid] = fmaxf(red[tid], red[tid + s]);
        __syncthreads();
    }
    float gmax = red[0];
    __syncthreads();

    float sm = 0.0f;
    for (int c = tid; c < Cn; c += 256) sm += expf(row[c] * invT - gmax);
    red[tid] = sm;
    __syncthreads();
    for (int s = 128; s > 0; s >>= 1) {
        if (tid < s) red[tid] += red[tid + s];
        __syncthreads();
    }
    float inv = 1.0f / red[0];

    for (int c = tid; c < CPAD; c += 256) {
        float v = (c < Cn) ? expf(row[c] * invT - gmax) * inv : 0.0f;
        Pt[(size_t)c * KMU + k] = (_Float16)v;
    }
}

// ---------------------------------------------------------------------------
// Fused: GEMM1 (x @ W_t^T via WMMA) -> sigmoid -> tree path products ->
// mu * softmax(tree_weights)[t], stored f16 as muw[b, t*64+l].
// Grid: (Bn/64, Tn). Block: 256 (8 waves); 64x64 tile, 2 WMMA tiles/wave.
// ---------------------------------------------------------------------------
__global__ void tree_kernel(const _Float16* __restrict__ xh,
                            const _Float16* __restrict__ swh,
                            const float* __restrict__ sbias,
                            const float* __restrict__ tw,
                            const float* __restrict__ logT,
                            _Float16* __restrict__ muw) {
    __shared__ float sh_S[64 * 64];
    __shared__ float sh_wt;
    __shared__ float sh_invT;

    int tid  = threadIdx.x;
    int wave = tid >> 5;
    int lane = tid & 31;
    int tr     = wave >> 1;          // tile row 0..3
    int tcBase = (wave & 1) * 2;     // tile cols {0,1} or {2,3}
    int m  = lane & 15;
    int hs = lane >> 4;
    int b0 = blockIdx.x * 64;
    int t  = blockIdx.y;

    if (tid == 0) {
        float temp = fminf(fmaxf(expf(logT[0]), 0.1f), 5.0f);
        float mxw = -INFINITY;
        #pragma unroll
        for (int i = 0; i < Tn; ++i) mxw = fmaxf(mxw, tw[i]);
        float s = 0.0f;
        #pragma unroll
        for (int i = 0; i < Tn; ++i) s += expf(tw[i] - mxw);
        sh_wt   = expf(tw[t] - mxw) / s;
        sh_invT = 1.0f / temp;
    }
    __syncthreads();

    const _Float16* Arow  = xh  + (size_t)(b0 + tr * 16 + m) * Dn;
    const _Float16* Brow0 = swh + ((size_t)t * NLEAF + tcBase * 16 + m) * Dn;
    const _Float16* Brow1 = Brow0 + 16 * Dn;

    v8f acc0 = {};
    v8f acc1 = {};
    for (int k0 = 0; k0 < Dn; k0 += 32) {
        v16h a  = load_frag16(Arow,  k0, hs);
        v16h bA = load_frag16(Brow0, k0, hs);
        v16h bB = load_frag16(Brow1, k0, hs);
        acc0 = wmma_f16(a, bA, acc0);
        acc1 = wmma_f16(a, bB, acc1);
    }

    // sigmoid((logits + bias)/temp) into LDS tile
    float invT = sh_invT;
    int nn0 = tcBase * 16 + m;
    int nn1 = nn0 + 16;
    float bv0 = (nn0 < NINT) ? sbias[t * NINT + nn0] : 0.0f;
    float bv1 = (nn1 < NINT) ? sbias[t * NINT + nn1] : 0.0f;
    #pragma unroll
    for (int r = 0; r < 8; ++r) {
        int mm = tr * 16 + r + 8 * hs;
        float l0 = (acc0[r] + bv0) * invT;
        float l1 = (acc1[r] + bv1) * invT;
        sh_S[mm * 64 + nn0] = 1.0f / (1.0f + expf(-l0));
        sh_S[mm * 64 + nn1] = 1.0f / (1.0f + expf(-l1));
    }
    __syncthreads();

    // mu expansion: thread -> (row 0..63, leaf-group 0..3 of 16 leaves)
    int row = tid >> 2;
    int lg  = tid & 3;
    const float* Sr = sh_S + row * 64;

    float mu = sh_wt;        // fold softmaxed tree weight
    int node = 0;
    #pragma unroll
    for (int d = 0; d < 2; ++d) {
        int bit = (lg >> (1 - d)) & 1;
        float s = Sr[node];
        mu *= bit ? s : (1.0f - s);
        node = 2 * node + 1 + bit;
    }
    float mv[16];
    mv[0] = mu;
    #pragma unroll
    for (int d = 0; d < 4; ++d) {
        int base = ((node + 1) << d) - 1;      // heap nodes of subtree at rel-depth d
        #pragma unroll 8
        for (int i = (1 << d) - 1; i >= 0; --i) {
            float s = Sr[base + i];
            mv[2 * i + 1] = mv[i] * s;
            mv[2 * i]     = mv[i] * (1.0f - s);
        }
    }

    union { v8h h[2]; _Float16 s16[16]; } o;
    #pragma unroll
    for (int j = 0; j < 16; ++j) o.s16[j] = (_Float16)mv[j];
    _Float16* dst = muw + (size_t)(b0 + row) * KMU + t * NLEAF + lg * 16;
    reinterpret_cast<v8h*>(dst)[0] = o.h[0];
    reinterpret_cast<v8h*>(dst)[1] = o.h[1];
}

// ---------------------------------------------------------------------------
// GEMM2: out[b,c] = sum_k muw[b,k] * Pt[c,k]; M=4096, N=1024(pad), K=2048.
// Grid: (64, 16). Block: 256 (8 waves), 64x64 tile, 2 WMMA tiles/wave.
// ---------------------------------------------------------------------------
__global__ void out_kernel(const _Float16* __restrict__ muw,
                           const _Float16* __restrict__ Pt,
                           float* __restrict__ out) {
    int tid  = threadIdx.x;
    int wave = tid >> 5;
    int lane = tid & 31;
    int tr     = wave >> 1;
    int tcBase = (wave & 1) * 2;
    int m  = lane & 15;
    int hs = lane >> 4;
    int b0 = blockIdx.x * 64;
    int c0 = blockIdx.y * 64;

    const _Float16* Arow  = muw + (size_t)(b0 + tr * 16 + m) * KMU;
    const _Float16* Brow0 = Pt  + (size_t)(c0 + tcBase * 16 + m) * KMU;
    const _Float16* Brow1 = Brow0 + (size_t)16 * KMU;

    v8f acc0 = {};
    v8f acc1 = {};
    for (int k0 = 0; k0 < KMU; k0 += 32) {
        v16h a  = load_frag16(Arow,  k0, hs);
        v16h bA = load_frag16(Brow0, k0, hs);
        v16h bB = load_frag16(Brow1, k0, hs);
        acc0 = wmma_f16(a, bA, acc0);
        acc1 = wmma_f16(a, bB, acc1);
    }

    int c_0 = c0 + tcBase * 16 + m;
    int c_1 = c_0 + 16;
    #pragma unroll
    for (int r = 0; r < 8; ++r) {
        int b = b0 + tr * 16 + r + 8 * hs;
        if (c_0 < Cn) out[(size_t)b * Cn + c_0] = acc0[r];
        if (c_1 < Cn) out[(size_t)b * Cn + c_1] = acc1[r];
    }
}

// ---------------------------------------------------------------------------
// Host launcher
// ---------------------------------------------------------------------------
extern "C" void kernel_launch(void* const* d_in, const int* in_sizes, int n_in,
                              void* d_out, int out_size, void* d_ws, size_t ws_size,
                              hipStream_t stream) {
    const float* x     = (const float*)d_in[0];
    const float* sw    = (const float*)d_in[1];
    const float* sb    = (const float*)d_in[2];
    const float* leafl = (const float*)d_in[3];
    const float* tw    = (const float*)d_in[4];
    const float* logT  = (const float*)d_in[5];
    float* out = (float*)d_out;

    char* ws = (char*)d_ws;
    const size_t sz_muw = (size_t)Bn * KMU * 2;        // 16 MiB
    const size_t sz_Pt  = (size_t)CPAD * KMU * 2;      //  4 MiB
    const size_t sz_xh  = (size_t)Bn * Dn * 2;         //  8 MiB
    _Float16* muw = (_Float16*)(ws);
    _Float16* Pt  = (_Float16*)(ws + sz_muw);
    _Float16* xh  = (_Float16*)(ws + sz_muw + sz_Pt);
    _Float16* swh = (_Float16*)(ws + sz_muw + sz_Pt + sz_xh);

    // 1) f32 -> f16 staging
    cvt_x_kernel<<<(Bn * Dn) / 256, 256, 0, stream>>>(x, xh);
    cvt_w_kernel<<<(Tn * NLEAF * Dn) / 256, 256, 0, stream>>>(sw, swh);

    // 2) leaf softmax -> transposed f16 Pt
    leaf_kernel<<<Tn * NLEAF, 256, 0, stream>>>(leafl, logT, Pt);

    // 3) fused GEMM1 + sigmoid + tree-path products (WMMA)
    tree_kernel<<<dim3(Bn / 64, Tn), 256, 0, stream>>>(xh, swh, sb, tw, logT, muw);

    // 4) GEMM2 -> output (WMMA)
    out_kernel<<<dim3(Bn / 64, CPAD / 64), 256, 0, stream>>>(muw, Pt, out);
}
